// RNN_Build_56882546868334
// MI455X (gfx1250) — matrix operand order repaired and checked by reading, not compiled, not run on hardware
//
#include <hip/hip_runtime.h>
#include <hip/hip_bf16.h>

#define SEQ     80
#define EMB     100
#define EMBP    128     // EMB padded to 4 K-chunks of 32
#define UNITS   64
#define BATCHN  8192
#define NWORDS  10000
#define WAVES   2       // waves per block
#define ROWS    16      // batch rows per wave (WMMA M)

typedef __attribute__((ext_vector_type(16))) __bf16 v16bf;
typedef __attribute__((ext_vector_type(8)))  __bf16 v8bf;
typedef __attribute__((ext_vector_type(8)))  float  v8f;

// ---- fast transcendentals on hardware units --------------------------------

#if __has_builtin(__builtin_amdgcn_exp2f)
#define EXP2F(x) __builtin_amdgcn_exp2f(x)
#else
#define EXP2F(x) exp2f(x)
#endif

#if __has_builtin(__builtin_amdgcn_rcpf)
#define RCPF(x) __builtin_amdgcn_rcpf(x)
#else
#define RCPF(x) (1.0f / (x))
#endif

__device__ __forceinline__ float fast_tanh(float x) {
  // tanh(x) = 1 - 2/(exp2(2*log2e*x) + 1); exact limits: +inf->1, -inf->-1
  float e = EXP2F(x * 2.885390081777927f);   // 2*log2(e)
  return 1.0f - 2.0f * RCPF(e + 1.0f);
}

__device__ __forceinline__ float fast_sigmoid(float x) {
  return RCPF(1.0f + EXP2F(x * -1.4426950408889634f));
}

// ---- fragment loaders (per CDNA5 ISA 16-bit WMMA layouts) -------------------

__device__ __forceinline__ v16bf combine(v8bf lo, v8bf hi) {
  v16bf f;
#pragma unroll
  for (int i = 0; i < 8; ++i) { f[i] = lo[i]; f[8 + i] = hi[i]; }
  return f;
}

// A-matrix 16xK fragment from row-major [16][stride] bf16 LDS buffer.
// lanes 0-15: row=lane, K = k0+{0..7, 16..23}; lanes 16-31: row=lane-16, +8.
__device__ __forceinline__ v16bf frag_a(const __bf16* base, int stride, int lane, int k0) {
  int r  = lane & 15;
  int kk = k0 + ((lane & 16) ? 8 : 0);
  const __bf16* p = base + r * stride + kk;
  return combine(*(const v8bf*)(p), *(const v8bf*)(p + 16));
}

// B-matrix Kx16 fragment from transposed [N][stride] bf16 LDS buffer.
// lanes 0-15: col=lane, K = k0+{0..7, 16..23}; lanes 16-31: col=lane-16, +8.
__device__ __forceinline__ v16bf frag_b(const __bf16* baseT, int stride, int lane, int k0) {
  int n  = lane & 15;
  int kk = k0 + ((lane & 16) ? 8 : 0);
  const __bf16* p = baseT + n * stride + kk;
  return combine(*(const v8bf*)(p), *(const v8bf*)(p + 16));
}

__device__ __forceinline__ v8f wmma_bf16(v16bf a, v16bf b, v8f c) {
  return __builtin_amdgcn_wmma_f32_16x16x32_bf16(
      /*neg_a=*/false, a, /*neg_b=*/false, b,
      /*c_mod=*/(short)0, c, /*reuse_a=*/false, /*reuse_b=*/false);
}

// ---- init kernel: emb f32 [10000][100] -> bf16 [10000][128] (zero padded) ---

__global__ void emb_cvt_kernel(const float* __restrict__ emb, __bf16* __restrict__ out) {
  int idx = blockIdx.x * blockDim.x + threadIdx.x;         // one element each
  if (idx >= NWORDS * EMBP) return;
  int w = idx >> 7, k = idx & (EMBP - 1);
  out[idx] = (__bf16)((k < EMB) ? emb[w * EMB + k] : 0.0f);
}

// ---- fused 2-layer SimpleRNN + sigmoid head ---------------------------------

template <bool BF16TAB>
__global__ __launch_bounds__(WAVES * 32)
void rnn2_fused_kernel(const int* __restrict__ tokens, const float* __restrict__ emb,
                       const __bf16* __restrict__ embbf,
                       const float* __restrict__ W1x, const float* __restrict__ W1h,
                       const float* __restrict__ b1,
                       const float* __restrict__ W2x, const float* __restrict__ W2h,
                       const float* __restrict__ b2,
                       const float* __restrict__ Wd, const float* __restrict__ bd,
                       float* __restrict__ out) {
  __shared__ __bf16 sW1x[UNITS][EMBP];    // W1x^T, zero-padded K   (16 KB)
  __shared__ __bf16 sW1h[UNITS][UNITS];   // W1h^T                  ( 8 KB)
  __shared__ __bf16 sW2x[UNITS][UNITS];   // W2x^T                  ( 8 KB)
  __shared__ __bf16 sW2h[UNITS][UNITS];   // W2h^T                  ( 8 KB)
  __shared__ __bf16 sH1[WAVES][ROWS][UNITS];  // per-wave h1 state  ( 4 KB)
  __shared__ __bf16 sH2[WAVES][ROWS][UNITS];  // per-wave h2 state  ( 4 KB)

  const int tid  = threadIdx.x;
  const int lane = tid & 31;
  const int wv   = tid >> 5;
  const int b0   = (blockIdx.x * WAVES + wv) * ROWS;

  // Stage weights: transpose to [N][K], convert f32 -> bf16, pad EMB->128.
  for (int idx = tid; idx < UNITS * EMBP; idx += WAVES * 32) {
    int n = idx >> 7, k = idx & (EMBP - 1);
    sW1x[n][k] = (__bf16)((k < EMB) ? W1x[k * UNITS + n] : 0.0f);
  }
  for (int idx = tid; idx < UNITS * UNITS; idx += WAVES * 32) {
    int n = idx >> 6, k = idx & (UNITS - 1);
    sW1h[n][k] = (__bf16)W1h[k * UNITS + n];
    sW2x[n][k] = (__bf16)W2x[k * UNITS + n];
    sW2h[n][k] = (__bf16)W2h[k * UNITS + n];
  }
  for (int idx = tid; idx < WAVES * ROWS * UNITS; idx += WAVES * 32) {
    (&sH1[0][0][0])[idx] = (__bf16)0.0f;
    (&sH2[0][0][0])[idx] = (__bf16)0.0f;
  }
  __syncthreads();   // cross-wave: the shared weight tiles and the sH1/sH2 zero-init

  const int m = lane & 15;                 // my M row / N col within a tile
  const long rowTok = (long)(b0 + m) * SEQ;

  // Bias pre-broadcast into WMMA C layout (element j -> row M, my col N).
  v8f biasC1[4], biasC2[4];
#pragma unroll
  for (int nt = 0; nt < 4; ++nt) {
    float v1 = b1[nt * 16 + m], v2 = b2[nt * 16 + m];
#pragma unroll
    for (int j = 0; j < 8; ++j) { biasC1[nt][j] = v1; biasC2[nt][j] = v2; }
  }

  const __bf16* h1b = &sH1[wv][0][0];
  const __bf16* h2b = &sH2[wv][0][0];

  // No in-loop barriers: sH1/sH2 are private to this wave; intra-wave LDS
  // store->load ordering is enforced by the in-order DS pipeline + s_wait_dscnt.
  for (int t = 0; t < SEQ; ++t) {
    // ---- gather x_t A-fragments (L2-resident table) ------------------------
    int tok = tokens[rowTok + t];
    v16bf ax[4];
    if constexpr (BF16TAB) {
      const __bf16* er = embbf + (long)tok * EMBP;   // pre-converted, padded
#pragma unroll
      for (int c = 0; c < 4; ++c) {
        int kk = c * 32 + ((lane & 16) ? 8 : 0);
        ax[c] = combine(*(const v8bf*)(er + kk), *(const v8bf*)(er + kk + 16));
      }
    } else {
      const float* er = emb + (long)tok * EMB;
#pragma unroll
      for (int c = 0; c < 4; ++c) {
        int kk = c * 32 + ((lane & 16) ? 8 : 0);
        v16bf f;
#pragma unroll
        for (int i = 0; i < 8; ++i) {
          int ka = kk + i, kb = kk + 16 + i;
          f[i]     = (__bf16)((ka < EMB) ? er[ka] : 0.0f);
          f[8 + i] = (__bf16)((kb < EMB) ? er[kb] : 0.0f);
        }
        ax[c] = f;
      }
    }

    // ---- layer 1: h1 = tanh(x@W1x + h1@W1h + b1) ---------------------------
    v8f acc1[4];
#pragma unroll
    for (int nt = 0; nt < 4; ++nt) {
      v8f acc = wmma_bf16(ax[0], frag_b(&sW1x[nt * 16][0], EMBP, lane, 0), biasC1[nt]);
#pragma unroll
      for (int c = 1; c < 4; ++c)
        acc = wmma_bf16(ax[c], frag_b(&sW1x[nt * 16][0], EMBP, lane, c * 32), acc);
#pragma unroll
      for (int c = 0; c < 2; ++c)
        acc = wmma_bf16(frag_a(h1b, UNITS, lane, c * 32),
                        frag_b(&sW1h[nt * 16][0], UNITS, lane, c * 32), acc);
      acc1[nt] = acc;
    }
    // D layout -> row-major bf16 state: lane l, vgpr j => M = j + (l>=16)*8, N = nt*16 + (l&15)
#pragma unroll
    for (int nt = 0; nt < 4; ++nt) {
      int n = nt * 16 + m;
#pragma unroll
      for (int j = 0; j < 8; ++j) {
        int mm = j + ((lane & 16) ? 8 : 0);
        sH1[wv][mm][n] = (__bf16)fast_tanh(acc1[nt][j]);
      }
    }

    // ---- layer 2: h2 = tanh(h1@W2x + h2@W2h + b2) --------------------------
    v8f acc2[4];
#pragma unroll
    for (int nt = 0; nt < 4; ++nt) {
      v8f acc = wmma_bf16(frag_a(h1b, UNITS, lane, 0),
                          frag_b(&sW2x[nt * 16][0], UNITS, lane, 0), biasC2[nt]);
      acc = wmma_bf16(frag_a(h2b, UNITS, lane, 0),
                      frag_b(&sW2h[nt * 16][0], UNITS, lane, 0), acc);
      acc = wmma_bf16(frag_a(h1b, UNITS, lane, 32),
                      frag_b(&sW2x[nt * 16][0], UNITS, lane, 32), acc);
      acc = wmma_bf16(frag_a(h2b, UNITS, lane, 32),
                      frag_b(&sW2h[nt * 16][0], UNITS, lane, 32), acc);
      acc2[nt] = acc;
    }
#pragma unroll
    for (int nt = 0; nt < 4; ++nt) {
      int n = nt * 16 + m;
#pragma unroll
      for (int j = 0; j < 8; ++j) {
        int mm = j + ((lane & 16) ? 8 : 0);
        sH2[wv][mm][n] = (__bf16)fast_tanh(acc2[nt][j]);
      }
    }
  }

  // ---- head: sigmoid(h2_last @ Wd + bd), one row per lane 0..15 ------------
  if (lane < 16) {
    float acc = bd[0];
#pragma unroll
    for (int k = 0; k < UNITS; ++k) acc += (float)sH2[wv][m][k] * Wd[k];
    out[b0 + m] = fast_sigmoid(acc);
  }
}

// ---- launcher ---------------------------------------------------------------

extern "C" void kernel_launch(void* const* d_in, const int* in_sizes, int n_in,
                              void* d_out, int out_size, void* d_ws, size_t ws_size,
                              hipStream_t stream) {
  (void)in_sizes; (void)n_in; (void)out_size;
  const int*   tokens = (const int*)d_in[0];
  const float* emb    = (const float*)d_in[1];
  const float* W1x    = (const float*)d_in[2];
  const float* W1h    = (const float*)d_in[3];
  const float* b1     = (const float*)d_in[4];
  const float* W2x    = (const float*)d_in[5];
  const float* W2h    = (const float*)d_in[6];
  const float* b2     = (const float*)d_in[7];
  const float* Wd     = (const float*)d_in[8];
  const float* bd     = (const float*)d_in[9];
  float* out = (float*)d_out;

  dim3 grid(BATCHN / (WAVES * ROWS));   // 256 blocks
  dim3 block(WAVES * 32);               // 64 threads = 2 waves

  const size_t need = (size_t)NWORDS * EMBP * sizeof(__bf16);   // 2.56 MB
  if (ws_size >= need) {
    __bf16* embbf = (__bf16*)d_ws;
    int total = NWORDS * EMBP;
    emb_cvt_kernel<<<(total + 255) / 256, 256, 0, stream>>>(emb, embbf);
    rnn2_fused_kernel<true><<<grid, block, 0, stream>>>(
        tokens, emb, embbf, W1x, W1h, b1, W2x, W2h, b2, Wd, bd, out);
  } else {
    rnn2_fused_kernel<false><<<grid, block, 0, stream>>>(
        tokens, emb, nullptr, W1x, W1h, b1, W2x, W2h, b2, Wd, bd, out);
  }
}
